// TSTiEncoder_25297357373599
// MI455X (gfx1250) — matrix operand order repaired and checked by reading, not compiled, not run on hardware
//
#include <hip/hip_runtime.h>
#include <hip/hip_bf16.h>
#include <math.h>

// ---------------------------------------------------------------------------
// TSTiEncoder (bidirectional Mamba encoder) for MI455X / gfx1250.
//  - GEMMs: v_wmma_f32_16x16x32_bf16, 32x32 tile/wave (4 accumulators),
//    weights pre-packed to WMMA-native bf16 layout (1 contiguous 32B
//    load per B fragment), compile-time-K fully unrolled inner loop
//  - selective scan: 64-chunk associative decomposition (3 phases),
//    B/C staged via global_load_async_to_lds_b32 double-buffering
//  - backward mamba runs direction-reversed conv/scan (no flips materialized)
// Workspace use: ~136 MB (< 192 MB L2).
// ---------------------------------------------------------------------------

typedef __attribute__((ext_vector_type(16))) __bf16 v16bf;
typedef __attribute__((ext_vector_type(8)))  float  v8f;

#define DMODEL 128
#define DSTATE 16
#define DTRANK 8
#define DFF    256
#define NB     8          // batch
#define NV     8          // nvars
#define PN     512        // patches
#define PL     16         // patch_len
#define LSEQ   (NV*PN+1)  // 4097
#define MROWS  (NB*LSEQ)  // 32776
#define NCHUNK 64
#define CHUNK  65         // ceil(4097/64)

__device__ __forceinline__ float sigm_(float x){ return 1.f/(1.f+expf(-x)); }

// ---- CDNA5 async global->LDS copy (ASYNCcnt-tracked) -----------------------
__device__ __forceinline__ void async_g2l_b32(const float* gsrc, float* lds_dst){
  unsigned long long ga = (unsigned long long)(uintptr_t)gsrc;
  unsigned int       la = (unsigned int)(uintptr_t)lds_dst;   // low 32b = LDS offset
  asm volatile("global_load_async_to_lds_b32 %0, %1, off"
               :: "v"(la), "v"(ga) : "memory");
}
__device__ __forceinline__ void wait_async0(){
  asm volatile("s_wait_asynccnt 0x0" ::: "memory");
}

// ------------------------- patch embedding ---------------------------------
__global__ void embed_kernel(const float* __restrict__ x, const float* __restrict__ view,
                             const float* __restrict__ Wp, const float* __restrict__ Wpb,
                             float* __restrict__ u){
  int idx = blockIdx.x*blockDim.x + threadIdx.x;
  if (idx >= MROWS*DMODEL) return;
  int d = idx & (DMODEL-1);
  int m = idx >> 7;
  int b = m / LSEQ, t = m % LSEQ;
  float acc;
  if (t == LSEQ-1){
    acc = view[b*DMODEL + d];
  } else {
    int n = t / PN, pi = t % PN;
    const float* xp = x + (((size_t)(b*NV+n)*PN + pi) * PL);
    acc = Wpb[d];
#pragma unroll
    for (int k=0;k<PL;++k) acc += xp[k]*Wp[k*DMODEL+d];
  }
  u[idx] = acc;
}

// ---- pack W (KxN f32, row-major) into WMMA-native bf16 B-fragment order ----
// P[((tn*KB + kb)*32 + lane)*16 + i] = W[(kb*32 + 16*(i>>3) + 8*(lane>>4) + (i&7))*N + tn*16 + (lane&15)]
// cols >= N zero-padded; tilesNp is rounded up to an even tile count.
__global__ void pack_w_kernel(const float* __restrict__ W, __bf16* __restrict__ P,
                              int K, int N, int tilesNp){
  int idx = blockIdx.x*blockDim.x + threadIdx.x;
  int KB = K >> 5;
  int total = tilesNp * KB * 512;
  if (idx >= total) return;
  int i    = idx & 15;
  int lane = (idx >> 4) & 31;
  int kb   = (idx >> 9) % KB;
  int tn   = (idx >> 9) / KB;
  int k    = (kb << 5) + ((i >> 3) << 4) + ((lane >> 4) << 3) + (i & 7);
  int col  = tn*16 + (lane & 15);
  float v  = (col < N) ? W[(size_t)k*N + col] : 0.f;
  P[idx]   = (__bf16)v;
}

// ---- A fragment: 16x32 bf16, lane m = l&15, K = (i>=8?16:0) + 8*(l>>4) + (i&7)
__device__ __forceinline__ v16bf load_a_frag(const float* ab){
  float4 a0 = *(const float4*)(ab);
  float4 a1 = *(const float4*)(ab + 4);
  float4 a2 = *(const float4*)(ab + 16);
  float4 a3 = *(const float4*)(ab + 20);
  v16bf f;
  f[0]=(__bf16)a0.x;  f[1]=(__bf16)a0.y;  f[2]=(__bf16)a0.z;  f[3]=(__bf16)a0.w;
  f[4]=(__bf16)a1.x;  f[5]=(__bf16)a1.y;  f[6]=(__bf16)a1.z;  f[7]=(__bf16)a1.w;
  f[8]=(__bf16)a2.x;  f[9]=(__bf16)a2.y;  f[10]=(__bf16)a2.z; f[11]=(__bf16)a2.w;
  f[12]=(__bf16)a3.x; f[13]=(__bf16)a3.y; f[14]=(__bf16)a3.z; f[15]=(__bf16)a3.w;
  return f;
}

__device__ __forceinline__ float epi_(float v, int flags){
  if (flags & 2){
    float in = 0.7978845608028654f*(v + 0.044715f*v*v*v);
    v = 0.5f*v*(1.f + tanhf(in));
  }
  return v;
}

// ------------------------- WMMA GEMM  C = act(A@W + bias) -------------------
// Wave computes a 32x32 tile: 4 accumulators, 2 A frags, 2 packed-B frags.
// flags: bit0 = accumulate into C, bit1 = tanh-GELU epilogue
template<int KT>
__global__ void gemm_kernel(const float* __restrict__ A, int lda,
                            const __bf16* __restrict__ Wpk,
                            const float* __restrict__ bias,
                            float* __restrict__ Cmat, int ldc,
                            int M, int N, int flags, int totalTiles){
  int wave = blockIdx.x*(blockDim.x>>5) + (threadIdx.x>>5);
  if (wave >= totalTiles) return;                 // wave-uniform branch
  constexpr int KB = KT >> 5;
  int lane = threadIdx.x & 31;
  int tilesN2 = (N+31)>>5;
  int tm2 = wave / tilesN2, tn2 = wave % tilesN2;
  int gl = lane >> 4;                             // lane group 0/1
  int ml = lane & 15;
  int r0 = tm2*32 + ml;
  int r1 = r0 + 16;
  int r0c = r0 < M ? r0 : M-1;                    // clamp, guard store only
  int r1c = r1 < M ? r1 : M-1;
  const float* Ap0 = A + (size_t)r0c*lda + (gl<<3);
  const float* Ap1 = A + (size_t)r1c*lda + (gl<<3);
  const __bf16* pb0 = Wpk + ((size_t)(2*tn2+0)*KB)*512 + lane*16;
  const __bf16* pb1 = Wpk + ((size_t)(2*tn2+1)*KB)*512 + lane*16;
  v8f acc00={}, acc01={}, acc10={}, acc11={};
#pragma unroll
  for (int kb=0; kb<KB; ++kb){
    v16bf a0 = load_a_frag(Ap0 + (kb<<5));
    v16bf a1 = load_a_frag(Ap1 + (kb<<5));
    v16bf b0 = *(const v16bf*)(pb0 + (size_t)kb*512);
    v16bf b1 = *(const v16bf*)(pb1 + (size_t)kb*512);
    acc00 = __builtin_amdgcn_wmma_f32_16x16x32_bf16(false, a0, false, b0,
                                                    (short)0, acc00, false, false);
    acc01 = __builtin_amdgcn_wmma_f32_16x16x32_bf16(false, a0, false, b1,
                                                    (short)0, acc01, false, false);
    acc10 = __builtin_amdgcn_wmma_f32_16x16x32_bf16(false, a1, false, b0,
                                                    (short)0, acc10, false, false);
    acc11 = __builtin_amdgcn_wmma_f32_16x16x32_bf16(false, a1, false, b1,
                                                    (short)0, acc11, false, false);
  }
  int c0 = tn2*32 + ml;
  int c1 = c0 + 16;
  float bv0 = bias ? bias[c0 < N ? c0 : 0] : 0.f;
  float bv1 = bias ? bias[c1 < N ? c1 : 0] : 0.f;
#pragma unroll
  for (int r=0;r<8;++r){
    int row0 = tm2*32 + r + (gl<<3);
    int row1 = row0 + 16;
    if (row0 < M){
      float* cr = Cmat + (size_t)row0*ldc;
      if (c0 < N){ float v = epi_(acc00[r]+bv0, flags); if (flags&1) cr[c0]+=v; else cr[c0]=v; }
      if (c1 < N){ float v = epi_(acc01[r]+bv1, flags); if (flags&1) cr[c1]+=v; else cr[c1]=v; }
    }
    if (row1 < M){
      float* cr = Cmat + (size_t)row1*ldc;
      if (c0 < N){ float v = epi_(acc10[r]+bv0, flags); if (flags&1) cr[c0]+=v; else cr[c0]=v; }
      if (c1 < N){ float v = epi_(acc11[r]+bv1, flags); if (flags&1) cr[c1]+=v; else cr[c1]=v; }
    }
  }
}

// ------------------- depthwise causal conv (k=2) + SiLU ---------------------
__global__ void conv_silu_kernel(const float* __restrict__ xz,   // M x 256, xi cols 0..127
                                 const float* __restrict__ cw,   // (128,2)
                                 const float* __restrict__ cb,
                                 float* __restrict__ xc, int dir){
  int idx = blockIdx.x*blockDim.x + threadIdx.x;
  if (idx >= MROWS*DMODEL) return;
  int d = idx & (DMODEL-1);
  int m = idx >> 7;
  int t = m % LSEQ;
  float xcur = xz[(size_t)m*(2*DMODEL) + d];
  float xnb  = 0.f;
  if (dir == 0){ if (t > 0)       xnb = xz[(size_t)(m-1)*(2*DMODEL) + d]; }
  else         { if (t < LSEQ-1)  xnb = xz[(size_t)(m+1)*(2*DMODEL) + d]; }
  float v = cw[d*2+0]*xnb + cw[d*2+1]*xcur + cb[d];
  xc[(size_t)m*DMODEL + d] = v * sigm_(v);
}

// ------------------- dt = softplus(dbl[:, :8] @ dt_w + dt_b) ----------------
__global__ void dt_kernel(const float* __restrict__ dbl,   // M x 40
                          const float* __restrict__ dtw,   // 8 x 128
                          const float* __restrict__ dtb,
                          float* __restrict__ dt){
  int idx = blockIdx.x*blockDim.x + threadIdx.x;
  if (idx >= MROWS*DMODEL) return;
  int d = idx & (DMODEL-1);
  int m = idx >> 7;
  const float* r = dbl + (size_t)m*40;
  float s = dtb[d];
#pragma unroll
  for (int k=0;k<DTRANK;++k) s += r[k]*dtw[k*DMODEL+d];
  dt[idx] = (s > 20.f) ? s : log1pf(expf(s));
}

// ------------------- scan phase A: per-chunk (prodA, accB) ------------------
// B vector staged via async global->LDS, double buffered: one barrier/step.
__global__ void scanA_kernel(const float* __restrict__ dt, const float* __restrict__ xc,
                             const float* __restrict__ dbl, const float* __restrict__ A_log,
                             float* __restrict__ cA, float* __restrict__ cB, int dir){
  int bc = blockIdx.x;
  int b = bc / NCHUNK, c = bc % NCHUNK;
  int d = threadIdx.x;
  __shared__ float sB[2][DSTATE];
  float Areg[DSTATE], ap[DSTATE], bp[DSTATE];
#pragma unroll
  for (int n=0;n<DSTATE;++n){ Areg[n] = -expf(A_log[d*DSTATE+n]); ap[n]=1.f; bp[n]=0.f; }
  int s0 = c*CHUNK;
  int s1 = s0 + CHUNK; if (s1 > LSEQ) s1 = LSEQ;
  {
    int t = dir ? (LSEQ-1-s0) : s0;
    int m = b*LSEQ + t;
    if (d < DSTATE) async_g2l_b32(&dbl[(size_t)m*40 + DTRANK + d], &sB[0][d]);
  }
  wait_async0();
  __syncthreads();
  int buf = 0;
  for (int s=s0; s<s1; ++s){
    int t = dir ? (LSEQ-1-s) : s;
    int m = b*LSEQ + t;
    if (s+1 < s1){                              // stage next step's B
      int t2 = dir ? (LSEQ-1-(s+1)) : (s+1);
      int m2 = b*LSEQ + t2;
      if (d < DSTATE) async_g2l_b32(&dbl[(size_t)m2*40 + DTRANK + d], &sB[buf^1][d]);
    }
    float dtv = dt[(size_t)m*DMODEL + d];
    float xcv = xc[(size_t)m*DMODEL + d];
    if (s+8 < s1){
      int tp = dir ? (LSEQ-1-(s+8)) : (s+8);
      size_t mp = (size_t)(b*LSEQ+tp)*DMODEL + d;
      __builtin_prefetch(&dt[mp],0,0);
      __builtin_prefetch(&xc[mp],0,0);
    }
#pragma unroll
    for (int n=0;n<DSTATE;++n){
      float dA = expf(dtv*Areg[n]);
      ap[n] *= dA;
      bp[n]  = dA*bp[n] + dtv*sB[buf][n]*xcv;
    }
    wait_async0();
    __syncthreads();
    buf ^= 1;
  }
  size_t base = ((size_t)(b*NCHUNK+c)*DMODEL + d)*DSTATE;
#pragma unroll
  for (int n=0;n<DSTATE;++n){ cA[base+n]=ap[n]; cB[base+n]=bp[n]; }
}

// ------------------- scan phase B: mid-scan over chunk states ---------------
__global__ void scanB_kernel(const float* __restrict__ cA, const float* __restrict__ cB,
                             float* __restrict__ hinit){
  int idx = blockIdx.x*blockDim.x + threadIdx.x;
  if (idx >= NB*DMODEL*DSTATE) return;
  int b  = idx / (DMODEL*DSTATE);
  int dn = idx % (DMODEL*DSTATE);
  float h = 0.f;
  for (int c=0;c<NCHUNK;++c){
    size_t base = ((size_t)(b*NCHUNK+c))*(DMODEL*DSTATE) + dn;
    hinit[base] = h;
    h = cA[base]*h + cB[base];
  }
}

// ------------------- scan phase C: replay + y + gate ------------------------
// B and C (32 contiguous floats: dbl cols 8..39) staged async, double buffered.
__global__ void scanC_kernel(const float* __restrict__ dt, const float* __restrict__ xc,
                             const float* __restrict__ dbl, const float* __restrict__ A_log,
                             const float* __restrict__ Dp,  const float* __restrict__ xz,
                             const float* __restrict__ hinit,
                             float* __restrict__ g, int dir){
  int bc = blockIdx.x;
  int b = bc / NCHUNK, c = bc % NCHUNK;
  int d = threadIdx.x;
  __shared__ float sBC[2][2*DSTATE];            // [0..15]=B, [16..31]=C
  float Areg[DSTATE], h[DSTATE];
  size_t hb = ((size_t)(b*NCHUNK+c)*DMODEL + d)*DSTATE;
#pragma unroll
  for (int n=0;n<DSTATE;++n){ Areg[n] = -expf(A_log[d*DSTATE+n]); h[n] = hinit[hb+n]; }
  float Dd = Dp[d];
  int s0 = c*CHUNK;
  int s1 = s0 + CHUNK; if (s1 > LSEQ) s1 = LSEQ;
  {
    int t = dir ? (LSEQ-1-s0) : s0;
    int m = b*LSEQ + t;
    if (d < 2*DSTATE) async_g2l_b32(&dbl[(size_t)m*40 + DTRANK + d], &sBC[0][d]);
  }
  wait_async0();
  __syncthreads();
  int buf = 0;
  for (int s=s0; s<s1; ++s){
    int t = dir ? (LSEQ-1-s) : s;
    int m = b*LSEQ + t;
    if (s+1 < s1){
      int t2 = dir ? (LSEQ-1-(s+1)) : (s+1);
      int m2 = b*LSEQ + t2;
      if (d < 2*DSTATE) async_g2l_b32(&dbl[(size_t)m2*40 + DTRANK + d], &sBC[buf^1][d]);
    }
    float dtv = dt[(size_t)m*DMODEL + d];
    float xcv = xc[(size_t)m*DMODEL + d];
    if (s+8 < s1){
      int tp = dir ? (LSEQ-1-(s+8)) : (s+8);
      size_t mp = (size_t)(b*LSEQ+tp)*DMODEL + d;
      __builtin_prefetch(&dt[mp],0,0);
      __builtin_prefetch(&xc[mp],0,0);
    }
    float acc = 0.f;
#pragma unroll
    for (int n=0;n<DSTATE;++n){
      float dA = expf(dtv*Areg[n]);
      h[n] = dA*h[n] + dtv*sBC[buf][n]*xcv;
      acc += h[n]*sBC[buf][DSTATE+n];
    }
    float y  = acc + xcv*Dd;
    float zv = xz[(size_t)m*(2*DMODEL) + DMODEL + d];
    g[(size_t)m*DMODEL + d] = y * (zv * sigm_(zv));
    wait_async0();
    __syncthreads();
    buf ^= 1;
  }
}

// ------------------- fused residual-add + LayerNorm -------------------------
__global__ void ln_kernel(const float* __restrict__ x, const float* __restrict__ res,
                          const float* __restrict__ gam, const float* __restrict__ bet,
                          float* __restrict__ out){
  int row = blockIdx.x;
  int d = threadIdx.x;
  __shared__ float red[DMODEL];
  float v = x[(size_t)row*DMODEL + d] + (res ? res[(size_t)row*DMODEL + d] : 0.f);
  red[d] = v; __syncthreads();
  for (int o=64;o>0;o>>=1){ if (d<o) red[d]+=red[d+o]; __syncthreads(); }
  float mu = red[0]*(1.f/DMODEL); __syncthreads();
  float dv = v - mu;
  red[d] = dv*dv; __syncthreads();
  for (int o=64;o>0;o>>=1){ if (d<o) red[d]+=red[d+o]; __syncthreads(); }
  float var = red[0]*(1.f/DMODEL);
  out[(size_t)row*DMODEL + d] = dv*rsqrtf(var + 1e-5f)*gam[d] + bet[d];
}

// ------------------- final (b,n,p,d) -> (b,n,d,p) transpose -----------------
__global__ void out_kernel(const float* __restrict__ u, float* __restrict__ out){
  int idx = blockIdx.x*blockDim.x + threadIdx.x;
  if (idx >= NB*NV*DMODEL*PN) return;
  int p = idx % PN;  int tmp = idx / PN;
  int d = tmp % DMODEL; tmp /= DMODEL;
  int n = tmp % NV;  int b = tmp / NV;
  int t = n*PN + p;
  out[idx] = u[((size_t)(b*LSEQ + t))*DMODEL + d];
}

// ---------------------------------------------------------------------------
static inline void launch_gemm(const float* A, int lda, const __bf16* Wpk,
                               const float* bias, float* C, int ldc,
                               int M, int N, int K, int flags, hipStream_t s){
  int tilesM2 = (M+31)>>5, tilesN2 = (N+31)>>5;
  int total   = tilesM2*tilesN2;
  int blocks  = (total + 7)/8;              // 8 waves (256 thr) per block
  if (K == 128)
    gemm_kernel<128><<<blocks, 256, 0, s>>>(A, lda, Wpk, bias, C, ldc, M, N, flags, total);
  else
    gemm_kernel<256><<<blocks, 256, 0, s>>>(A, lda, Wpk, bias, C, ldc, M, N, flags, total);
}

extern "C" void kernel_launch(void* const* d_in, const int* in_sizes, int n_in,
                              void* d_out, int out_size, void* d_ws, size_t ws_size,
                              hipStream_t stream){
  (void)in_sizes; (void)n_in; (void)out_size; (void)ws_size;
  // Input flattening (setup_inputs insertion order, depth-first):
  // 0:x 1:view 2:W_P_w 3:W_P_b
  // per layer (26 tensors starting at 4 + 26*layer):
  //   mf: +0 in_w +1 conv_w +2 conv_b +3 x_w +4 dt_w +5 dt_b +6 A_log +7 D +8 out_w
  //   mb: +9..+17 (same order)
  //   +18 ln1_g +19 ln1_b +20 w1 +21 b1 +22 w2 +23 b2 +24 ln2_g +25 ln2_b
  // 56: lnf_g  57: lnf_b
  const float* x    = (const float*)d_in[0];
  const float* view = (const float*)d_in[1];
  const float* Wp   = (const float*)d_in[2];
  const float* Wpb  = (const float*)d_in[3];
  const float* lnfg = (const float*)d_in[56];
  const float* lnfb = (const float*)d_in[57];

  float* ws   = (float*)d_ws;
  float* u    = ws;                                   // M*128
  float* t256 = u    + (size_t)MROWS*DMODEL;          // M*256 (xz / ffn hidden)
  float* xc   = t256 + (size_t)MROWS*2*DMODEL;        // M*128
  float* dbl  = xc   + (size_t)MROWS*DMODEL;          // M*40
  float* dtb  = dbl  + (size_t)MROWS*40;              // M*128
  float* gbuf = dtb  + (size_t)MROWS*DMODEL;          // M*128
  float* macc = gbuf + (size_t)MROWS*DMODEL;          // M*128
  float* cA   = macc + (size_t)MROWS*DMODEL;          // 8*64*128*16
  float* cB   = cA   + (size_t)NB*NCHUNK*DMODEL*DSTATE;
  float* hin  = cB   + (size_t)NB*NCHUNK*DMODEL*DSTATE;
  __bf16* wp  = (__bf16*)(hin + (size_t)NB*NCHUNK*DMODEL*DSTATE);

  const int thr = 256;
  const int ewb = (MROWS*DMODEL + thr-1)/thr;

  // ---- pack all GEMM weights to WMMA-native bf16 layout (tiny, ~1MB) ------
  __bf16* pk_in[2][2]; __bf16* pk_x[2][2]; __bf16* pk_out[2][2];
  __bf16* pk_w1[2];    __bf16* pk_w2[2];
  auto packW = [&](const float* W, int K, int N) -> __bf16* {
    int tilesNp = ((N+31)/32)*2;
    int KB = K/32;
    size_t elems = (size_t)tilesNp*KB*512;
    __bf16* dst = wp; wp += elems;
    pack_w_kernel<<<(int)((elems+255)/256), 256, 0, stream>>>(W, dst, K, N, tilesNp);
    return dst;
  };
  for (int layer=0; layer<2; ++layer){
    int base = 4 + layer*26;
    for (int dir=0; dir<2; ++dir){
      int mb = base + dir*9;
      pk_in [layer][dir] = packW((const float*)d_in[mb+0], DMODEL, 2*DMODEL);
      pk_x  [layer][dir] = packW((const float*)d_in[mb+3], DMODEL, DTRANK+2*DSTATE);
      pk_out[layer][dir] = packW((const float*)d_in[mb+8], DMODEL, DMODEL);
    }
    pk_w1[layer] = packW((const float*)d_in[base+20], DMODEL, DFF);
    pk_w2[layer] = packW((const float*)d_in[base+22], DFF, DMODEL);
  }

  embed_kernel<<<ewb, thr, 0, stream>>>(x, view, Wp, Wpb, u);

  for (int layer=0; layer<2; ++layer){
    int base = 4 + layer*26;
    for (int dir=0; dir<2; ++dir){
      int mb = base + dir*9;
      const float* cw    = (const float*)d_in[mb+1];
      const float* cb    = (const float*)d_in[mb+2];
      const float* dt_w  = (const float*)d_in[mb+4];
      const float* dt_bp = (const float*)d_in[mb+5];
      const float* A_log = (const float*)d_in[mb+6];
      const float* Dp    = (const float*)d_in[mb+7];

      launch_gemm(u, DMODEL, pk_in[layer][dir], nullptr, t256, 2*DMODEL,
                  MROWS, 2*DMODEL, DMODEL, 0, stream);                 // in_proj
      conv_silu_kernel<<<ewb, thr, 0, stream>>>(t256, cw, cb, xc, dir);
      launch_gemm(xc, DMODEL, pk_x[layer][dir], nullptr, dbl, 40,
                  MROWS, 40, DMODEL, 0, stream);                       // x_proj
      dt_kernel<<<ewb, thr, 0, stream>>>(dbl, dt_w, dt_bp, dtb);
      scanA_kernel<<<NB*NCHUNK, DMODEL, 0, stream>>>(dtb, xc, dbl, A_log, cA, cB, dir);
      scanB_kernel<<<(NB*DMODEL*DSTATE + thr-1)/thr, thr, 0, stream>>>(cA, cB, hin);
      scanC_kernel<<<NB*NCHUNK, DMODEL, 0, stream>>>(dtb, xc, dbl, A_log, Dp, t256,
                                                     hin, gbuf, dir);
      launch_gemm(gbuf, DMODEL, pk_out[layer][dir], nullptr, macc, DMODEL,
                  MROWS, DMODEL, DMODEL, dir ? 1 : 0, stream);         // out_proj (+=)
    }
    const float* ln1g = (const float*)d_in[base+18];
    const float* ln1b = (const float*)d_in[base+19];
    const float* b1   = (const float*)d_in[base+21];
    const float* b2   = (const float*)d_in[base+23];
    const float* ln2g = (const float*)d_in[base+24];
    const float* ln2b = (const float*)d_in[base+25];

    ln_kernel<<<MROWS, DMODEL, 0, stream>>>(u, macc, ln1g, ln1b, u);
    launch_gemm(u, DMODEL, pk_w1[layer], b1, t256, DFF,
                MROWS, DFF, DMODEL, 2, stream);                        // GELU
    launch_gemm(t256, DFF, pk_w2[layer], b2, macc, DMODEL,
                MROWS, DMODEL, DFF, 0, stream);
    ln_kernel<<<MROWS, DMODEL, 0, stream>>>(u, macc, ln2g, ln2b, u);
  }
  ln_kernel<<<MROWS, DMODEL, 0, stream>>>(u, nullptr, lnfg, lnfb, u);

  int owb = (NB*NV*DMODEL*PN + thr-1)/thr;
  out_kernel<<<owb, thr, 0, stream>>>(u, (float*)d_out);
}
